// MultiViewEncoder_45260365365597
// MI455X (gfx1250) — matrix operand order repaired and checked by reading, not compiled, not run on hardware
//
#include <hip/hip_runtime.h>

// ---------------------------------------------------------------------------
// Problem constants (from reference)
// ---------------------------------------------------------------------------
#define N_TOT   262144
#define G_TOT   65536
#define D_IN    512
#define D_MID   256
#define D_GAT   128
#define D_OUT   256
#define BN_EPS  1e-5f
#define L2_EPS  1e-9f

typedef __attribute__((ext_vector_type(16))) __bf16    v16bf;
typedef __attribute__((ext_vector_type(8)))  float     v8f;
typedef __attribute__((ext_vector_type(4)))  unsigned  v4u;
typedef __attribute__((ext_vector_type(8)))  int       v8i;
typedef __attribute__((ext_vector_type(4)))  int       v4i;

#if __has_builtin(__builtin_amdgcn_tensor_load_to_lds)
#define USE_TDM 1
#endif

// ---------------------------------------------------------------------------
// WMMA fragment helpers (CDNA5 §7.12.2 layouts, wave32)
//   A 16x32 bf16: lane<16 -> M=lane, K = {k0..k0+7, k0+16..k0+23}
//                 lane>=16 -> M=lane-16, K = {k0+8..k0+15, k0+24..k0+31}
//   B chunk in LDS, K-transposed wTl[n][k], stride 40 bf16 (80B rows):
//                 lane<16 -> N=n0+lane, K={0..7,16..23}; lane>=16 -> K={8..15,24..31}
//   C/D 16x16 f32: lane<16 -> N=lane, rows m0..m0+7 in v[0..7]
//                  lane>=16 -> N=lane-16, rows m0+8..m0+15
// ---------------------------------------------------------------------------
__device__ __forceinline__ v16bf load_frag_a(const __bf16* base, int stride,
                                             int m0, int k0, int lane) {
    const __bf16* p = base + (size_t)(m0 + (lane & 15)) * stride + k0 + ((lane >> 4) << 3);
    union { uint4 u[2]; v16bf v; } c;
    c.u[0] = *(const uint4*)(p);
    c.u[1] = *(const uint4*)(p + 16);
    return c.v;
}

__device__ __forceinline__ v16bf load_frag_b(const __bf16* wTl, int n0, int lane) {
    const __bf16* p = wTl + (size_t)(n0 + (lane & 15)) * 40 + ((lane >> 4) << 3);
    union { uint4 u[2]; v16bf v; } c;
    c.u[0] = *(const uint4*)(p);
    c.u[1] = *(const uint4*)(p + 16);
    return c.v;
}

__device__ __forceinline__ v8f wmma_bf16(v16bf a, v16bf b, v8f c) {
    return __builtin_amdgcn_wmma_f32_16x16x32_bf16(false, a, false, b,
                                                   (short)0, c, false, false);
}

__device__ __forceinline__ v8f zero8() {
    return (v8f){0.f, 0.f, 0.f, 0.f, 0.f, 0.f, 0.f, 0.f};
}

// ---------------------------------------------------------------------------
// LDS byte-offset of a __shared__ pointer (addrspacecast generic -> LDS)
// ---------------------------------------------------------------------------
__device__ __forceinline__ unsigned lds_offset(const void* p) {
    return (unsigned)(unsigned long long)
        (__attribute__((address_space(3))) const void*)p;
}

// ---------------------------------------------------------------------------
// Stage one K-chunk (32 K x NC cols) of a K-transposed bf16 weight matrix
// wTg[n*Ktot + k] into LDS laid out wTl[n*40 + k] (stride 40 = 32 + 4dw pad).
//
// TDM path: 2D tile, dim0 = 32 contiguous K elements (64B), dim1 = NC rows,
// row stride Ktot elements; LDS padding 4 dwords per 16 dwords reproduces the
// stride-40 layout.  Issued by wave 0 only; tracked with TENSORcnt.
// This toolchain exposes the 6-arg builtin:
//   (uint32x4 g0, int32x8 g1, int32x4 g2, int32x4 g3, int32x8, i32 cpol)
// ---------------------------------------------------------------------------
__device__ __forceinline__ void stage_chunk(__bf16* wTl, const __bf16* wTg,
                                            int Ktot, int NC, int kc,
                                            int wv, int t) {
#ifdef USE_TDM
    if (wv == 0) {
        unsigned long long ga = (unsigned long long)(wTg + (size_t)kc * 32);
        v4u g0;
        g0[0] = 1u;                                   // count=1, user descriptor
        g0[1] = lds_offset(wTl);                      // lds_addr (bytes)
        g0[2] = (unsigned)(ga & 0xffffffffu);         // global_addr[31:0]
        g0[3] = (unsigned)((ga >> 32) & 0x1ffffffu)   // global_addr[56:32]
              | (2u << 30);                           // type=2 (image)
        v8i g1;
        g1[0] = (1 << 16)       // data_size = 2B
              | (1 << 20)       // pad_enable
              | (3 << 22)       // pad_interval: every 16 dwords
              | (3 << 25);      // pad_amount:  4 dwords
        g1[1] = (32 & 0xffff) << 16;          // tensor_dim0 = 32 (tile-local)
        g1[2] = (NC & 0xffff) << 16;          // tensor_dim1 = NC
        g1[3] = (32 << 16);                   // tile_dim0 = 32
        g1[4] = (NC & 0xffff);                // tile_dim1 = NC, tile_dim2 = 0
        g1[5] = Ktot;                         // tensor_dim0_stride (elements)
        g1[6] = 0;
        g1[7] = 0;
        v4i z4 = {0, 0, 0, 0};
        v8i z8 = {0, 0, 0, 0, 0, 0, 0, 0};
        __builtin_amdgcn_tensor_load_to_lds(g0, g1, z4, z4, z8, 0);
    }
#else
    const __bf16* src = wTg + (size_t)kc * 32;
    int total = (NC * 32) >> 3;               // # of 8-bf16 packets
    for (int i = t; i < total; i += 256) {
        int lin = i << 3;
        int n = lin >> 5, k8 = lin & 31;      // k8 in {0,8,16,24}
        *(uint4*)&wTl[n * 40 + k8] = *(const uint4*)(src + (size_t)n * Ktot + k8);
    }
#endif
}

__device__ __forceinline__ void stage_wait(int wv) {
#ifdef USE_TDM
    if (wv == 0) {
#if __has_builtin(__builtin_amdgcn_s_wait_tensorcnt)
        __builtin_amdgcn_s_wait_tensorcnt(0);
#else
        asm volatile("s_wait_tensorcnt 0x0" ::: "memory");
#endif
    }
#endif
    __syncthreads();
}

// ---------------------------------------------------------------------------
// Prep kernels
// ---------------------------------------------------------------------------
// Transpose fp32 [K x N] -> bf16 [N x K] (one-time, tiny matrices).
__global__ void k_cvt_T(const float* __restrict__ src, __bf16* __restrict__ dst,
                        int K, int N) {
    int i = blockIdx.x * 256 + threadIdx.x;
    if (i < K * N) {
        int k = i / N, n = i % N;
        dst[(size_t)n * K + k] = (__bf16)src[i];
    }
}

// Fold BatchNorm(eval) into per-column scale/bias applied after x@w2.
__global__ void k_scale_bias(const float* __restrict__ bn_g, const float* __restrict__ bn_b,
                             const float* __restrict__ bn_m, const float* __restrict__ bn_v,
                             const float* __restrict__ b2,
                             float* __restrict__ scale, float* __restrict__ bias) {
    int i = threadIdx.x;
    if (i < D_MID) {
        float s = bn_g[i] * rsqrtf(bn_v[i] + BN_EPS);
        scale[i] = s;
        bias[i] = (b2[i] - bn_m[i]) * s + bn_b[i];
    }
}

// seg_ids sorted, all segments non-empty -> record first row of each segment.
__global__ void k_segstart(const int* __restrict__ seg, int* __restrict__ segstart) {
    int r = blockIdx.x * 256 + threadIdx.x;
    if (r < N_TOT) {
        if (r == 0 || seg[r] != seg[r - 1]) segstart[seg[r]] = r;
    }
}

// ---------------------------------------------------------------------------
// Kernel 1: fused MLP.  64 rows / block, 256 threads (8 waves).
//   h1 = relu(scale*(x@w2)+bias)     (bf16 in LDS)
//   h2 = relu(h1@w3+b3)              (fp32 -> global, bf16 in LDS)
//   g  = h2@gw+gb                    (fp32 -> global)
//   e  = g . ga                      (fp32 -> global, via ds_add_f32)
// Weight chunks double-buffered; TDM DMA overlaps WMMA.
// ---------------------------------------------------------------------------
struct Smem1 {
    __bf16 xs[64 * 520];      // x tile (1040B rows, 16B aligned, bank-spread)
    __bf16 wT[2][256 * 40];   // double-buffered K-chunk, K-transposed
    __bf16 h1s[64 * 264];
    __bf16 h2s[64 * 136];
    float  es[64];
};

__global__ __launch_bounds__(256)
void k_fused_mlp(const float* __restrict__ x,
                 const __bf16* __restrict__ w2t, const __bf16* __restrict__ w3t,
                 const __bf16* __restrict__ gwt,
                 const float* __restrict__ scale, const float* __restrict__ bias,
                 const float* __restrict__ b3, const float* __restrict__ gb,
                 const float* __restrict__ ga,
                 float* __restrict__ h2g, float* __restrict__ gg,
                 float* __restrict__ eg) {
    __shared__ Smem1 sm;
    const int t = threadIdx.x;
    const int wv = t >> 5, lane = t & 31;
    const int rt = (wv & 3) * 16;          // row-tile base
    const int half = wv >> 2;              // column-group selector
    const int row0 = blockIdx.x * 64;

    if (t < 64) sm.es[t] = 0.f;

    // ---- stage x tile [64 x 512] fp32 -> bf16 LDS -------------------------
    {
        const float4* xp = (const float4*)(x + (size_t)row0 * D_IN);
#pragma unroll
        for (int i = 0; i < 32; i++) {
            int idx4 = t + 256 * i;              // 8192 float4 total
            int r = idx4 >> 7, c = (idx4 & 127) << 2;
            float4 v = xp[idx4];
            union { __bf16 h[4]; uint2 u; } pk;
            pk.h[0] = (__bf16)v.x; pk.h[1] = (__bf16)v.y;
            pk.h[2] = (__bf16)v.z; pk.h[3] = (__bf16)v.w;
            *(uint2*)&sm.xs[r * 520 + c] = pk.u;
        }
    }

    // ---- GEMM1: [64x512] @ [512x256], 16 K-chunks, double-buffered --------
    v8f acc[8];
#pragma unroll
    for (int j = 0; j < 8; j++) acc[j] = zero8();

    stage_chunk(sm.wT[0], w2t, D_IN, D_MID, 0, wv, t);
    stage_wait(wv);
    for (int kc = 0; kc < 16; kc++) {
        if (kc + 1 < 16) stage_chunk(sm.wT[(kc + 1) & 1], w2t, D_IN, D_MID, kc + 1, wv, t);
        v16bf a = load_frag_a(sm.xs, 520, rt, kc * 32, lane);
#pragma unroll
        for (int j = 0; j < 8; j++) {
            v16bf b = load_frag_b(sm.wT[kc & 1], (half * 8 + j) * 16, lane);
            acc[j] = wmma_bf16(a, b, acc[j]);
        }
        stage_wait(wv);
    }
    // epilogue: folded BN + ReLU -> h1s (bf16)
#pragma unroll
    for (int j = 0; j < 8; j++) {
        int n = (half * 8 + j) * 16 + (lane & 15);
        float sc = scale[n], bs = bias[n];
#pragma unroll
        for (int r = 0; r < 8; r++) {
            int m = rt + ((lane >> 4) << 3) + r;
            float v = fmaxf(acc[j][r] * sc + bs, 0.f);
            sm.h1s[m * 264 + n] = (__bf16)v;
        }
    }

    // ---- GEMM2: [64x256] @ [256x128], 8 K-chunks --------------------------
    v8f acc2[4];
#pragma unroll
    for (int j = 0; j < 4; j++) acc2[j] = zero8();
    stage_chunk(sm.wT[0], w3t, D_MID, D_GAT, 0, wv, t);
    stage_wait(wv);   // also publishes h1s
    for (int kc = 0; kc < 8; kc++) {
        if (kc + 1 < 8) stage_chunk(sm.wT[(kc + 1) & 1], w3t, D_MID, D_GAT, kc + 1, wv, t);
        v16bf a = load_frag_a(sm.h1s, 264, rt, kc * 32, lane);
#pragma unroll
        for (int j = 0; j < 4; j++) {
            v16bf b = load_frag_b(sm.wT[kc & 1], (half * 4 + j) * 16, lane);
            acc2[j] = wmma_bf16(a, b, acc2[j]);
        }
        stage_wait(wv);
    }
#pragma unroll
    for (int j = 0; j < 4; j++) {
        int n = (half * 4 + j) * 16 + (lane & 15);
        float bb = b3[n];
#pragma unroll
        for (int r = 0; r < 8; r++) {
            int m = rt + ((lane >> 4) << 3) + r;
            float v = fmaxf(acc2[j][r] + bb, 0.f);
            h2g[(size_t)(row0 + m) * D_GAT + n] = v;
            sm.h2s[m * 136 + n] = (__bf16)v;
        }
    }

    // ---- GEMM3: [64x128] @ [128x128], 4 K-chunks; g + e -------------------
    v8f acc3[4];
#pragma unroll
    for (int j = 0; j < 4; j++) acc3[j] = zero8();
    stage_chunk(sm.wT[0], gwt, D_GAT, D_GAT, 0, wv, t);
    stage_wait(wv);   // also publishes h2s
    for (int kc = 0; kc < 4; kc++) {
        if (kc + 1 < 4) stage_chunk(sm.wT[(kc + 1) & 1], gwt, D_GAT, D_GAT, kc + 1, wv, t);
        v16bf a = load_frag_a(sm.h2s, 136, rt, kc * 32, lane);
#pragma unroll
        for (int j = 0; j < 4; j++) {
            v16bf b = load_frag_b(sm.wT[kc & 1], (half * 4 + j) * 16, lane);
            acc3[j] = wmma_bf16(a, b, acc3[j]);
        }
        stage_wait(wv);
    }
#pragma unroll
    for (int j = 0; j < 4; j++) {
        int n = (half * 4 + j) * 16 + (lane & 15);
        float bb = gb[n], gaf = ga[n];
#pragma unroll
        for (int r = 0; r < 8; r++) {
            int m = rt + ((lane >> 4) << 3) + r;
            float v = acc3[j][r] + bb;
            gg[(size_t)(row0 + m) * D_GAT + n] = v;
            atomicAdd(&sm.es[m], gaf * v);      // ds_add_f32
        }
    }
    __syncthreads();
    if (t < 64) eg[row0 + t] = sm.es[t];
}

// ---------------------------------------------------------------------------
// Kernel 2: one wave per segment (sorted, contiguous, non-empty).
//   segment softmax over e, attention pool of g, residual+relu with h2,
//   per-dim segment max -> obj[G,128].  Lane owns dims {4*lane..4*lane+3}
//   so all row traffic is b128.
// ---------------------------------------------------------------------------
__global__ __launch_bounds__(256)
void k_segment(const float* __restrict__ eg, const float* __restrict__ gg,
               const float* __restrict__ h2g, const int* __restrict__ segstart,
               float* __restrict__ obj) {
    int sid = blockIdx.x * 8 + (threadIdx.x >> 5);
    int lane = threadIdx.x & 31;
    int start = segstart[sid];
    int end = (sid + 1 < G_TOT) ? segstart[sid + 1] : N_TOT;

    float m = -1e30f;
    for (int r = start + lane; r < end; r += 32) m = fmaxf(m, eg[r]);
#pragma unroll
    for (int o = 16; o > 0; o >>= 1) m = fmaxf(m, __shfl_xor(m, o, 32));
    float den = 0.f;
    for (int r = start + lane; r < end; r += 32) den += __expf(eg[r] - m);
#pragma unroll
    for (int o = 16; o > 0; o >>= 1) den += __shfl_xor(den, o, 32);
    float inv = 1.f / den;

    float4 p = {0.f, 0.f, 0.f, 0.f};
    for (int r = start; r < end; r++) {
        float a = __expf(eg[r] - m) * inv;
        float4 gv = *(const float4*)(gg + (size_t)r * D_GAT + lane * 4);
        p.x += a * gv.x; p.y += a * gv.y; p.z += a * gv.z; p.w += a * gv.w;
    }
    float4 mx = {-1e30f, -1e30f, -1e30f, -1e30f};
    for (int r = start; r < end; r++) {
        float4 hv = *(const float4*)(h2g + (size_t)r * D_GAT + lane * 4);
        mx.x = fmaxf(mx.x, fmaxf(hv.x + p.x, 0.f));
        mx.y = fmaxf(mx.y, fmaxf(hv.y + p.y, 0.f));
        mx.z = fmaxf(mx.z, fmaxf(hv.z + p.z, 0.f));
        mx.w = fmaxf(mx.w, fmaxf(hv.w + p.w, 0.f));
    }
    *(float4*)(obj + (size_t)sid * D_GAT + lane * 4) = mx;
}

// ---------------------------------------------------------------------------
// Kernel 3: projection [G x 128] @ [128 x 256] + bias, then row L2-norm.
// ---------------------------------------------------------------------------
struct Smem3 {
    __bf16 objs[64 * 136];
    __bf16 wT[2][256 * 40];
    float  outs[64 * 260];
    float  nrm[64];
};

__global__ __launch_bounds__(256)
void k_project(const float* __restrict__ obj, const __bf16* __restrict__ pwt,
               const float* __restrict__ pb, float* __restrict__ out) {
    __shared__ Smem3 sm;
    const int t = threadIdx.x;
    const int wv = t >> 5, lane = t & 31;
    const int rt = (wv & 3) * 16;
    const int half = wv >> 2;
    const int row0 = blockIdx.x * 64;

    if (t < 64) sm.nrm[t] = 0.f;

    // stage obj tile [64 x 128] fp32 -> bf16
    {
        const float4* op = (const float4*)(obj + (size_t)row0 * D_GAT);
#pragma unroll
        for (int i = 0; i < 8; i++) {
            int idx4 = t + 256 * i;
            int r = idx4 >> 5, c = (idx4 & 31) << 2;
            float4 v = op[idx4];
            union { __bf16 h[4]; uint2 u; } pk;
            pk.h[0] = (__bf16)v.x; pk.h[1] = (__bf16)v.y;
            pk.h[2] = (__bf16)v.z; pk.h[3] = (__bf16)v.w;
            *(uint2*)&sm.objs[r * 136 + c] = pk.u;
        }
    }

    v8f acc[8];
#pragma unroll
    for (int j = 0; j < 8; j++) acc[j] = zero8();
    stage_chunk(sm.wT[0], pwt, D_GAT, D_OUT, 0, wv, t);
    stage_wait(wv);
    for (int kc = 0; kc < 4; kc++) {
        if (kc + 1 < 4) stage_chunk(sm.wT[(kc + 1) & 1], pwt, D_GAT, D_OUT, kc + 1, wv, t);
        v16bf a = load_frag_a(sm.objs, 136, rt, kc * 32, lane);
#pragma unroll
        for (int j = 0; j < 8; j++) {
            v16bf b = load_frag_b(sm.wT[kc & 1], (half * 8 + j) * 16, lane);
            acc[j] = wmma_bf16(a, b, acc[j]);
        }
        stage_wait(wv);
    }
#pragma unroll
    for (int j = 0; j < 8; j++) {
        int n = (half * 8 + j) * 16 + (lane & 15);
        float bb = pb[n];
#pragma unroll
        for (int r = 0; r < 8; r++) {
            int m = rt + ((lane >> 4) << 3) + r;
            float v = acc[j][r] + bb;
            sm.outs[m * 260 + n] = v;
            atomicAdd(&sm.nrm[m], v * v);        // ds_add_f32
        }
    }
    __syncthreads();
#pragma unroll
    for (int i = 0; i < 64; i++) {
        int lin = t + 256 * i;                   // 16384 elements
        int r = lin >> 8, c = lin & 255;
        float inv = 1.f / (sqrtf(sm.nrm[r]) + L2_EPS);
        out[(size_t)(row0 + r) * D_OUT + c] = sm.outs[r * 260 + c] * inv;
    }
}

// ---------------------------------------------------------------------------
// Host-side launcher
// ---------------------------------------------------------------------------
extern "C" void kernel_launch(void* const* d_in, const int* in_sizes, int n_in,
                              void* d_out, int out_size, void* d_ws, size_t ws_size,
                              hipStream_t stream) {
    (void)in_sizes; (void)n_in; (void)out_size; (void)ws_size;
    const float* x    = (const float*)d_in[0];
    const int*   seg  = (const int*)  d_in[1];
    const float* w2   = (const float*)d_in[2];
    const float* b2   = (const float*)d_in[3];
    const float* bn_g = (const float*)d_in[4];
    const float* bn_b = (const float*)d_in[5];
    const float* bn_m = (const float*)d_in[6];
    const float* bn_v = (const float*)d_in[7];
    const float* w3   = (const float*)d_in[8];
    const float* b3   = (const float*)d_in[9];
    const float* gw   = (const float*)d_in[10];
    const float* gb   = (const float*)d_in[11];
    const float* ga   = (const float*)d_in[12];
    const float* pw   = (const float*)d_in[13];
    const float* pb   = (const float*)d_in[14];
    float* out = (float*)d_out;

    // workspace carve-out (256B aligned)
    char* ws = (char*)d_ws;
    size_t off = 0;
    auto carve = [&](size_t bytes) {
        size_t o = off;
        off = (off + bytes + 255) & ~(size_t)255;
        return (void*)(ws + o);
    };
    __bf16* w2t = (__bf16*)carve((size_t)D_IN  * D_MID * 2);  // [256][512]
    __bf16* w3t = (__bf16*)carve((size_t)D_MID * D_GAT * 2);  // [128][256]
    __bf16* gwt = (__bf16*)carve((size_t)D_GAT * D_GAT * 2);  // [128][128]
    __bf16* pwt = (__bf16*)carve((size_t)D_GAT * D_OUT * 2);  // [256][128]
    float*  scale    = (float*)carve(D_MID * 4);
    float*  bias     = (float*)carve(D_MID * 4);
    int*    segstart = (int*)  carve((size_t)G_TOT * 4);
    float*  h2g      = (float*)carve((size_t)N_TOT * D_GAT * 4);
    float*  gg       = (float*)carve((size_t)N_TOT * D_GAT * 4);
    float*  eg       = (float*)carve((size_t)N_TOT * 4);
    float*  obj      = (float*)carve((size_t)G_TOT * D_GAT * 4);

    // prep: K-transpose weights to bf16 [N][K]
    k_cvt_T<<<(D_IN * D_MID + 255) / 256, 256, 0, stream>>>(w2, w2t, D_IN, D_MID);
    k_cvt_T<<<(D_MID * D_GAT + 255) / 256, 256, 0, stream>>>(w3, w3t, D_MID, D_GAT);
    k_cvt_T<<<(D_GAT * D_GAT + 255) / 256, 256, 0, stream>>>(gw, gwt, D_GAT, D_GAT);
    k_cvt_T<<<(D_GAT * D_OUT + 255) / 256, 256, 0, stream>>>(pw, pwt, D_GAT, D_OUT);
    k_scale_bias<<<1, 256, 0, stream>>>(bn_g, bn_b, bn_m, bn_v, b2, scale, bias);
    k_segstart<<<N_TOT / 256, 256, 0, stream>>>(seg, segstart);

    // fused MLP (WMMA bf16 + TDM weight staging)
    k_fused_mlp<<<N_TOT / 64, 256, 0, stream>>>(x, w2t, w3t, gwt, scale, bias,
                                                b3, gb, ga, h2g, gg, eg);
    // ragged attention pool + residual + segment max
    k_segment<<<G_TOT / 8, 256, 0, stream>>>(eg, gg, h2g, segstart, obj);
    // projection + L2 norm (WMMA bf16 + TDM weight staging)
    k_project<<<G_TOT / 64, 256, 0, stream>>>(obj, pwt, pb, out);
}
